// DoubleAttention_7713761264209
// MI455X (gfx1250) — matrix operand order repaired and checked by reading, not compiled, not run on hardware
//
#include <hip/hip_runtime.h>
#include <math.h>

// ---------------- problem constants ----------------
constexpr int BATCH = 4;
constexpr int CCH   = 64;              // input channels C
constexpr int NSP   = 48 * 48 * 48;    // 110592 spatial positions
constexpr int CM    = 64;              // c_m
constexpr int CN    = 32;              // c_n

constexpr int TN  = 128;               // spatial tile per block (8 waves x 16 cols)
constexpr int NTB = NSP / TN;          // 864 tiles per batch
constexpr int XP  = 72;                // x-tile LDS pitch (64+8 bf16, 144B rows)
constexpr int AP  = 136;               // 128-wide tile LDS pitch (272B rows)
constexpr int VP  = 40;                // eV LDS pitch (32+8 bf16, 80B rows)

// ---------------- workspace layout (floats) ----------------
constexpr int WS_MAXB  = 0;
constexpr int WS_MAXV  = WS_MAXB + BATCH * CN;
constexpr int WS_SUMB  = WS_MAXV + BATCH * CN;
constexpr int WS_SUMV  = WS_SUMB + BATCH * CN;
constexpr int WS_GD    = WS_SUMV + BATCH * CN;           // gd numerator [B][CM][CN]
constexpr int WS_ISV   = WS_GD   + BATCH * CM * CN;      // 1/sumV      [B][CN]
constexpr int WS_ZEND  = WS_GD   + BATCH * CM * CN;      // zero range end (sums..gd)
// Pre-formatted bf16 WMMA fragments (2 bf16 per float slot):
constexpr int WS_WFRAG = WS_ISV  + BATCH * CN;           // 16 frags * 32 lanes * 16 bf16
constexpr int WS_MFRAG = WS_WFRAG + (16 * 32 * 16) / 2;  // BATCH*4 frags * 32 * 16 bf16
constexpr int WS_END   = WS_MFRAG + (BATCH * 4 * 32 * 16) / 2;

typedef __attribute__((ext_vector_type(16))) __bf16 v16bf;
typedef __attribute__((ext_vector_type(8)))  __bf16 v8bf;
typedef __attribute__((ext_vector_type(8)))  float  v8f;

__device__ __forceinline__ v8f wmma_bf16(v16bf a, v16bf b, v8f c) {
  // D(16x16 f32) = A(16x32 bf16) * B(32x16 bf16) + C
  return __builtin_amdgcn_wmma_f32_16x16x32_bf16(false, a, false, b, (short)0, c,
                                                 false, false);
}

__device__ __forceinline__ v16bf cat8(v8bf a, v8bf b) {
  return __builtin_shufflevector(a, b, 0, 1, 2, 3, 4, 5, 6, 7, 8, 9, 10, 11, 12, 13, 14, 15);
}
__device__ __forceinline__ v8bf lo8(v16bf a) {
  return __builtin_shufflevector(a, a, 0, 1, 2, 3, 4, 5, 6, 7);
}
__device__ __forceinline__ v8bf hi8(v16bf a) {
  return __builtin_shufflevector(a, a, 8, 9, 10, 11, 12, 13, 14, 15);
}

// B fragment: Bmat[K][col]; LDS matrix stored [col][K] with given pitch.
// lane(r,hi): element j<8 = Bmat[k0 + hi*8 + j][col0 + r], j>=8 = Bmat[k0 + 16 + hi*8 + (j-8)][col0 + r]
__device__ __forceinline__ v16bf bfrag_lds(const __bf16* Ml, int pitch, int col0, int k0) {
  int lane = threadIdx.x & 31;
  int r = lane & 15, hi = lane >> 4;
  const __bf16* p = Ml + (col0 + r) * pitch + k0 + hi * 8;
  return cat8(*(const v8bf*)p, *(const v8bf*)(p + 16));
}

// A fragment from bf16 LDS stored [row][K]; runs [hi*8,+8) and [16+hi*8,+8)
__device__ __forceinline__ v16bf afrag_lds(const __bf16* Ml, int pitch, int row0, int k0) {
  int lane = threadIdx.x & 31;
  int r = lane & 15, hi = lane >> 4;
  const __bf16* p = Ml + (row0 + r) * pitch + k0 + hi * 8;
  return cat8(*(const v8bf*)p, *(const v8bf*)(p + 16));
}

// A fragment built from fp32 row-major [row][ld] (used only in prep kernels)
__device__ __forceinline__ v16bf afrag_f32(const float* W, int ld, int row0, int k0) {
  int lane = threadIdx.x & 31;
  int r = lane & 15, hi = lane >> 4;
  const float* p = W + (row0 + r) * ld + k0 + hi * 8;
  v16bf a;
#pragma unroll
  for (int j = 0; j < 8; ++j) {
    a[j]     = (__bf16)p[j];
    a[j + 8] = (__bf16)p[j + 16];
  }
  return a;
}

// Load a pre-formatted fragment: 32B per lane, fully coalesced, L2-hot.
__device__ __forceinline__ v16bf load_frag(const __bf16* base, int f) {
  const __bf16* p = base + ((f * 32 + (threadIdx.x & 31)) * 16);
  return cat8(*(const v8bf*)p, *(const v8bf*)(p + 8));
}
__device__ __forceinline__ void store_frag(__bf16* base, int f, v16bf a) {
  __bf16* p = base + ((f * 32 + (threadIdx.x & 31)) * 16);
  *(v8bf*)p       = lo8(a);
  *(v8bf*)(p + 8) = hi8(a);
}

// Straight-line scan of 32 contiguous bf16 in LDS (4 aligned 16B loads).
__device__ __forceinline__ float rowscan_max(const __bf16* p) {
  float m = -INFINITY;
#pragma unroll
  for (int q = 0; q < 4; ++q) {
    v8bf v = *(const v8bf*)(p + 8 * q);
#pragma unroll
    for (int j = 0; j < 8; ++j) m = fmaxf(m, (float)v[j]);
  }
  return m;
}
__device__ __forceinline__ float rowscan_sum(const __bf16* p) {
  float s = 0.f;
#pragma unroll
  for (int q = 0; q < 4; ++q) {
    v8bf v = *(const v8bf*)(p + 8 * q);
#pragma unroll
    for (int j = 0; j < 8; ++j) s += (float)v[j];
  }
  return s;
}

// Coalesced load of 64xTN fp32 x-tile -> bf16 LDS stored [n][c] (pitch XP)
__device__ __forceinline__ void load_xtile(const float* xb, int n0, __bf16* xs) {
  int t = threadIdx.x;
#pragma unroll
  for (int i = 0; i < 8; ++i) {
    int id = t + 256 * i;            // 0..2047 float4 ids
    int c  = id >> 5;                // 0..63
    int n4 = (id & 31) << 2;         // 0..124
    float4 v = *(const float4*)(xb + (size_t)c * NSP + n0 + n4);
    xs[(n4 + 0) * XP + c] = (__bf16)v.x;
    xs[(n4 + 1) * XP + c] = (__bf16)v.y;
    xs[(n4 + 2) * XP + c] = (__bf16)v.z;
    xs[(n4 + 3) * XP + c] = (__bf16)v.w;
  }
}

// ---------------- K0: reset accumulators (every launch; graph-safe) -------
__global__ void k_init(float* __restrict__ ws) {
  int t = blockIdx.x * blockDim.x + threadIdx.x;
  if (t < WS_SUMB) ws[t] = -INFINITY;        // maxB, maxV
  else if (t < WS_ZEND) ws[t] = 0.0f;        // sumB, sumV, gd
}

// ---------------- Kp: pre-format weight fragments (once per launch) -------
// Fragment ids f = rg*2 + kb:  rg 0-3 -> wA rows, rg 4-5 -> wB, rg 6-7 -> wV
__global__ __launch_bounds__(512) void k_prep(const float* __restrict__ wA,
                                              const float* __restrict__ wB,
                                              const float* __restrict__ wV,
                                              float* __restrict__ ws) {
  int f = threadIdx.x >> 5;                  // 0..15
  int rg = f >> 1, kb = f & 1;
  const float* W;
  int row0;
  if (rg < 4)      { W = wA; row0 = rg * 16; }
  else if (rg < 6) { W = wB; row0 = (rg - 4) * 16; }
  else             { W = wV; row0 = (rg - 6) * 16; }
  v16bf a = afrag_f32(W, CCH, row0, kb * 32);
  store_frag((__bf16*)(ws + WS_WFRAG), f, a);
}

// ---------------- K1: row maxes of Braw=wB@x, Vraw=wV@x -------------------
__global__ __launch_bounds__(256) void k_pass1(const float* __restrict__ x,
                                               float* __restrict__ ws) {
  __shared__ __attribute__((aligned(16))) __bf16 xs[TN * XP];
  __shared__ __attribute__((aligned(16))) __bf16 pv[64 * AP];  // 0-31 Braw, 32-63 Vraw
  int b = blockIdx.y, n0 = blockIdx.x * TN;
  load_xtile(x + (size_t)b * CCH * NSP, n0, xs);
  __syncthreads();

  int lane = threadIdx.x & 31, r = lane & 15, hi = lane >> 4;
  int cg = threadIdx.x >> 5;                        // wave id == column group
  int ncol = cg * 16 + r;
  const __bf16* wf = (const __bf16*)(ws + WS_WFRAG);
  v16bf xb0 = bfrag_lds(xs, XP, cg * 16, 0);
  v16bf xb1 = bfrag_lds(xs, XP, cg * 16, 32);

#pragma unroll
  for (int rg = 0; rg < 4; ++rg) {                  // rg 0-1: wB, 2-3: wV
    int fb = (4 + rg) * 2;
    v8f d = {};
    d = wmma_bf16(load_frag(wf, fb), xb0, d);
    d = wmma_bf16(load_frag(wf, fb + 1), xb1, d);
    int rbase = ((rg < 2) ? 0 : 32) + (rg & 1) * 16 + hi * 8;
#pragma unroll
    for (int v = 0; v < 8; ++v) pv[(rbase + v) * AP + ncol] = (__bf16)d[v];
  }
  __syncthreads();

  // 4 threads per row, each scans 32 contiguous values; one atomic each.
  int row = threadIdx.x >> 2, seg = threadIdx.x & 3;
  float m = rowscan_max(pv + row * AP + seg * 32);
  if (row < CN) atomicMax(ws + WS_MAXB + b * CN + row, m);
  else          atomicMax(ws + WS_MAXV + b * CN + (row - CN), m);
}

// ---------------- K2: A, eB, partial gd_num, exp-sums ---------------------
__global__ __launch_bounds__(256) void k_pass2(const float* __restrict__ x,
                                               const float* __restrict__ bA,
                                               float* __restrict__ ws) {
  __shared__ __attribute__((aligned(16))) __bf16 xs[TN * XP];
  __shared__ __attribute__((aligned(16))) __bf16 Asl[CM * AP];   // A  [m][n]
  __shared__ __attribute__((aligned(16))) __bf16 eBs[CN * AP];   // eB [k][n]
  __shared__ __attribute__((aligned(16))) __bf16 eVt[CN * AP];   // eV [k][n] (sum only)
  int b = blockIdx.y, n0 = blockIdx.x * TN;
  load_xtile(x + (size_t)b * CCH * NSP, n0, xs);
  __syncthreads();

  int lane = threadIdx.x & 31, r = lane & 15, hi = lane >> 4;
  int cg = threadIdx.x >> 5;
  int ncol = cg * 16 + r;
  const float* maxBp = ws + WS_MAXB + b * CN;
  const float* maxVp = ws + WS_MAXV + b * CN;
  const __bf16* wf = (const __bf16*)(ws + WS_WFRAG);
  v16bf xb0 = bfrag_lds(xs, XP, cg * 16, 0);
  v16bf xb1 = bfrag_lds(xs, XP, cg * 16, 32);

#pragma unroll
  for (int rg = 0; rg < 8; ++rg) {
    v8f d = {};
    d = wmma_bf16(load_frag(wf, rg * 2), xb0, d);
    d = wmma_bf16(load_frag(wf, rg * 2 + 1), xb1, d);
    int row0 = (rg < 4) ? rg * 16 : ((rg < 6) ? (rg - 4) * 16 : (rg - 6) * 16);
#pragma unroll
    for (int v = 0; v < 8; ++v) {
      int rr = row0 + hi * 8 + v;
      if (rg < 4) {
        Asl[rr * AP + ncol] = (__bf16)(d[v] + bA[rr]);          // A tile (+bias)
      } else if (rg < 6) {
        eBs[rr * AP + ncol] = (__bf16)__expf(d[v] - maxBp[rr]); // eB tile
      } else {
        eVt[rr * AP + ncol] = (__bf16)__expf(d[v] - maxVp[rr]); // eV tile (sums only)
      }
    }
  }
  __syncthreads();

  // exp-sums: 4 threads per row, straight-line scans + one atomic each.
  int row = threadIdx.x >> 2, seg = threadIdx.x & 3;
  if (row < CN) atomicAdd(ws + WS_SUMB + b * CN + row,
                          rowscan_sum(eBs + row * AP + seg * 32));
  else          atomicAdd(ws + WS_SUMV + b * CN + (row - CN),
                          rowscan_sum(eVt + (row - CN) * AP + seg * 32));

  // gd_num partial: (64 x TN) * (TN x 32), K dim = spatial tile
  int mg = cg & 3, kg = cg >> 2;
  v8f g = {};
#pragma unroll
  for (int s8 = 0; s8 < TN / 32; ++s8) {
    g = wmma_bf16(afrag_lds(Asl, AP, mg * 16, s8 * 32),
                  bfrag_lds(eBs, AP, kg * 16, s8 * 32), g);
  }
#pragma unroll
  for (int v = 0; v < 8; ++v) {
    int m = mg * 16 + hi * 8 + v;
    int k = kg * 16 + r;
    atomicAdd(ws + WS_GD + (b * CM + m) * CN + k, g[v]);
  }
}

// -------- K3: finalize M = wR @ (gd/sumB) and emit M as bf16 fragments ----
__global__ __launch_bounds__(256) void k_pass3(const float* __restrict__ wR,
                                               float* __restrict__ ws) {
  __shared__ float gds[CM * CN];
  __shared__ float msh[CCH * CN];
  int b = blockIdx.x, t = threadIdx.x;
  for (int i = t; i < CM * CN; i += 256) {
    int k = i & (CN - 1);
    gds[i] = ws[WS_GD + b * CM * CN + i] / ws[WS_SUMB + b * CN + k];
  }
  __syncthreads();
  for (int i = t; i < CCH * CN; i += 256) {
    int c = i / CN, k = i & (CN - 1);
    float acc = 0.f;
#pragma unroll 8
    for (int m = 0; m < CM; ++m) acc += wR[c * CM + m] * gds[m * CN + k];
    msh[i] = acc;
  }
  __syncthreads();

  int wv = t >> 5, lane = t & 31;
  if (wv < 4) {                                     // 4 fragments per batch
    int r = lane & 15, hi = lane >> 4;
    v16bf a;
#pragma unroll
    for (int j = 0; j < 8; ++j) {
      a[j]     = (__bf16)msh[(wv * 16 + r) * CN + hi * 8 + j];
      a[j + 8] = (__bf16)msh[(wv * 16 + r) * CN + 16 + hi * 8 + j];
    }
    store_frag((__bf16*)(ws + WS_MFRAG), b * 4 + wv, a);
  }
  if (t < CN) ws[WS_ISV + b * CN + t] = 1.0f / ws[WS_SUMV + b * CN + t];
}

// ---------------- K4: out = M @ softmax(wV@x) + bR ------------------------
__global__ __launch_bounds__(256) void k_pass4(const float* __restrict__ x,
                                               const float* __restrict__ bR,
                                               const float* __restrict__ ws,
                                               float* __restrict__ out) {
  __shared__ __attribute__((aligned(16))) __bf16 xs[TN * XP];
  __shared__ __attribute__((aligned(16))) __bf16 eVs[TN * VP];   // [n][k]
  int b = blockIdx.y, n0 = blockIdx.x * TN;
  load_xtile(x + (size_t)b * CCH * NSP, n0, xs);
  __syncthreads();

  int lane = threadIdx.x & 31, r = lane & 15, hi = lane >> 4;
  int cg = threadIdx.x >> 5;
  const float* maxVp = ws + WS_MAXV + b * CN;
  const float* isvp  = ws + WS_ISV  + b * CN;
  const __bf16* wf = (const __bf16*)(ws + WS_WFRAG);
  const __bf16* mf = (const __bf16*)(ws + WS_MFRAG);
  v16bf xb0 = bfrag_lds(xs, XP, cg * 16, 0);
  v16bf xb1 = bfrag_lds(xs, XP, cg * 16, 32);

#pragma unroll
  for (int rg = 0; rg < 2; ++rg) {                  // Vraw = wV @ x
    int fb = (6 + rg) * 2;
    v8f d = {};
    d = wmma_bf16(load_frag(wf, fb), xb0, d);
    d = wmma_bf16(load_frag(wf, fb + 1), xb1, d);
#pragma unroll
    for (int v = 0; v < 8; ++v) {
      int k = rg * 16 + hi * 8 + v;
      float e = __expf(d[v] - maxVp[k]) * isvp[k];
      eVs[(cg * 16 + r) * VP + k] = (__bf16)e;
    }
  }
  __syncthreads();

  v16bf vb = bfrag_lds(eVs, VP, cg * 16, 0);       // B[K=k][col=n]
  float* ob = out + (size_t)b * CCH * NSP + n0 + cg * 16 + r;
#pragma unroll
  for (int rg = 0; rg < 4; ++rg) {
    v8f d = {};
    d = wmma_bf16(load_frag(mf, b * 4 + rg), vb, d);   // K=32: one WMMA
#pragma unroll
    for (int v = 0; v < 8; ++v) {
      int c = rg * 16 + hi * 8 + v;
      // Output has no reuse: non-temporal store keeps x resident in the
      // 192MB L2 across the three streaming passes (x = 113MB fits).
      __builtin_nontemporal_store(d[v] + bR[c], ob + (size_t)c * NSP);
    }
  }
}

// ---------------- host-side launch ----------------------------------------
extern "C" void kernel_launch(void* const* d_in, const int* in_sizes, int n_in,
                              void* d_out, int out_size, void* d_ws, size_t ws_size,
                              hipStream_t stream) {
  (void)in_sizes; (void)n_in; (void)out_size; (void)ws_size;
  const float* x  = (const float*)d_in[0];
  const float* wA = (const float*)d_in[1];
  const float* bA = (const float*)d_in[2];
  const float* wB = (const float*)d_in[3];
  // d_in[4] = bB: constant along the softmax axis -> cancels exactly; unused
  const float* wV = (const float*)d_in[5];
  // d_in[6] = bV: cancels likewise; unused
  const float* wR = (const float*)d_in[7];
  const float* bR = (const float*)d_in[8];
  float* out = (float*)d_out;
  float* ws  = (float*)d_ws;

  k_init<<<dim3((WS_ZEND + 255) / 256), 256, 0, stream>>>(ws);
  k_prep<<<dim3(1), 512, 0, stream>>>(wA, wB, wV, ws);
  k_pass1<<<dim3(NTB, BATCH), 256, 0, stream>>>(x, ws);
  k_pass2<<<dim3(NTB, BATCH), 256, 0, stream>>>(x, bA, ws);
  k_pass3<<<dim3(BATCH), 256, 0, stream>>>(wR, ws);
  k_pass4<<<dim3(NTB, BATCH), 256, 0, stream>>>(x, bR, ws, out);
}
